// ComformerLayer_10788957847843
// MI455X (gfx1250) — matrix operand order repaired and checked by reading, not compiled, not run on hardware
//
#include <hip/hip_runtime.h>
#include <hip/hip_bf16.h>

#define D   128
#define LDW 136   // LDS row stride (ushort): 272B rows keep 16B alignment + bank skew

typedef __attribute__((ext_vector_type(16))) __bf16 bf16x16;
typedef __attribute__((ext_vector_type(8)))  float  f32x8;

union FragU { uint4 q[2]; bf16x16 v; };

__device__ __forceinline__ unsigned short f2bf(float x) {
    unsigned int u = __float_as_uint(x);
    unsigned int r = (u + 0x7FFFu + ((u >> 16) & 1u)) >> 16;
    return (unsigned short)r;
}

__device__ __forceinline__ float silu_f(float x) {
    return x / (1.0f + __expf(-x));
}

// Position of logical weight element (n = output col, k = reduction idx) inside the
// WMMA-fragment-packed weight image: tiles of 16 cols x 32 k, each lane owning a
// contiguous 32B run (this code's contiguous k order, same as A: lanes 0-15 -> K 0..15, lanes 16-31 -> K 16..31).
__device__ __forceinline__ size_t frag_index(int n, int k) {
    int t    = n >> 4;                       // 16-col tile
    int kk   = k >> 5;                       // 32-k step
    int lane = (((k >> 4) & 1) << 4) | (n & 15);
    int e    = k & 15;
    return ((size_t)(((t * 4 + kk) * 32) + lane)) * 16 + e;
}

// ---- one wave: 16x64 output tile = 4 accumulators; A from LDS, B straight from L2 ----
__device__ __forceinline__ void wmma_quad(const unsigned short* __restrict__ A,
                                          const unsigned short* __restrict__ Tp,
                                          int rowg, int tbase, int lane, f32x8 acc[4]) {
    #pragma unroll
    for (int j = 0; j < 4; ++j) acc[j] = f32x8{};
    const int ar    = rowg + (lane & 15);
    const int akoff = (lane >> 4) << 4;      // A: lanes16-31 hold K 16..31 run
    #pragma unroll
    for (int kk = 0; kk < 4; ++kk) {
        FragU fa;
        fa.q[0] = *(const uint4*)(A + ar * LDW + kk * 32 + akoff);
        fa.q[1] = *(const uint4*)(A + ar * LDW + kk * 32 + 8 + akoff);
        #pragma unroll
        for (int j = 0; j < 4; ++j) {
            const uint4* p = (const uint4*)(Tp + (size_t)(((tbase + j) * 4 + kk) * 32 + lane) * 16);
            FragU fb;
            fb.q[0] = p[0];
            fb.q[1] = p[1];
            acc[j] = __builtin_amdgcn_wmma_f32_16x16x32_bf16(false, fa.v, false, fb.v,
                                                             (short)0, acc[j], false, false);
        }
    }
}

// ================= prep kernels (tiny): fold weights, emit fragment-packed bf16 =================

// Tcomb[m] (m=0..5): fragment-packed bf16 of
//   m0: Wk@k1_w[0::3]  m1: Wk@k1_w[1::3]  m2: We@k1_w[2::3]
//   m3: Wv@v1_w[0::3]  m4: Wv@v1_w[1::3]  m5: We@v1_w[2::3]
__global__ void prep_combined_kernel(const float* __restrict__ Wk, const float* __restrict__ Wv,
                                     const float* __restrict__ We, const float* __restrict__ k1w,
                                     const float* __restrict__ v1w, unsigned short* __restrict__ Tcomb) {
    int m = blockIdx.x, i = blockIdx.y, j = threadIdx.x;   // i = k index, j = n index
    const float* S  = (m == 2 || m == 5) ? We : (m < 2 ? Wk : Wv);
    const float* W1 = (m < 3) ? k1w : v1w;
    int sel = m % 3;
    float acc = 0.0f;
    for (int t = 0; t < D; ++t)
        acc += S[i * D + t] * W1[(3 * t + sel) * D + j];
    Tcomb[(size_t)m * D * D + frag_index(j, i)] = f2bf(acc);
}

// Ttr[m] (m=0..2): fragment-packed bf16 of Wq, k2_w, v2_w
__global__ void prep_transpose_kernel(const float* __restrict__ Wq, const float* __restrict__ k2w,
                                      const float* __restrict__ v2w, unsigned short* __restrict__ Ttr) {
    int m = blockIdx.x, i = blockIdx.y, j = threadIdx.x;
    const float* S = (m == 0) ? Wq : (m == 1) ? k2w : v2w;
    Ttr[(size_t)m * D * D + frag_index(j, i)] = f2bf(S[i * D + j]);
}

__global__ void prep_bias_kernel(const float* __restrict__ bk, const float* __restrict__ be,
                                 const float* __restrict__ k1w, const float* __restrict__ k1b,
                                 const float* __restrict__ bv, const float* __restrict__ v1w,
                                 const float* __restrict__ v1b,
                                 float* __restrict__ bias_hk, float* __restrict__ bias_hv) {
    int j = threadIdx.x;
    float ak = k1b[j], av = v1b[j];
    for (int t = 0; t < D; ++t) {
        ak += bk[t] * (k1w[(3 * t) * D + j] + k1w[(3 * t + 1) * D + j]) + be[t] * k1w[(3 * t + 2) * D + j];
        av += bv[t] * (v1w[(3 * t) * D + j] + v1w[(3 * t + 1) * D + j]) + be[t] * v1w[(3 * t + 2) * D + j];
    }
    bias_hk[j] = ak;
    bias_hv[j] = av;
}

// ================= atom-stage GEMM: Y[M,128] = X[M,128] @ W (+bias), 64-row tiles =================
__global__ __launch_bounds__(256) void gemm_proj_kernel(const float* __restrict__ X, int M,
                                                        const unsigned short* __restrict__ Tp,
                                                        const float* __restrict__ bias,
                                                        float* __restrict__ Y) {
    __shared__ unsigned short sA[64][LDW];
    const int tid   = threadIdx.x;
    const int lane  = tid & 31;
    const int wave  = tid >> 5;
    const int rowg  = (wave & 3) << 4;       // 4 row groups of 16
    const int tbase = (wave >> 2) * 4;       // 2 col groups of 64 (4 n-tiles)
    const int colg  = tbase * 16;
    const long long x0 = (long long)blockIdx.x * 64;

    #pragma unroll
    for (int i = tid; i < 64 * 32; i += 256) {   // 64 rows x 32 float4
        int r = i >> 5, c = (i & 31) << 2;
        long long row = x0 + r; if (row >= M) row = M - 1;
        float4 f = *(const float4*)(X + row * D + c);
        uint2 p;
        p.x = ((unsigned int)f2bf(f.y) << 16) | f2bf(f.x);
        p.y = ((unsigned int)f2bf(f.w) << 16) | f2bf(f.z);
        *(uint2*)&sA[r][c] = p;
    }
    __syncthreads();

    f32x8 acc[4];
    wmma_quad(&sA[0][0], Tp, rowg, tbase, lane, acc);

    const int mb = rowg + ((lane >> 4) << 3);
    #pragma unroll
    for (int j = 0; j < 4; ++j) {
        int n = colg + j * 16 + (lane & 15);
        float bj = bias ? bias[n] : 0.0f;
        #pragma unroll
        for (int r = 0; r < 8; ++r) {
            long long row = x0 + mb + r;
            if (row < M) Y[row * D + n] = acc[j][r] + bj;
        }
    }
}

// ================= fused edge kernel: 64 edges/block =================
// h_k = E@Wke + Ka[src]+Kb[dst]+bias ; k = silu(h_k)@k2 + k2_b ; a = Qa[dst]*k/sqrt(d)
// h_v likewise ; v stored
__global__ __launch_bounds__(256) void edge_fused_kernel(
    const float* __restrict__ edgef, const int* __restrict__ eidx, int E,
    const float* __restrict__ Qa, const float* __restrict__ Ka, const float* __restrict__ Kb,
    const float* __restrict__ Va, const float* __restrict__ Vb,
    const float* __restrict__ bias_hk, const float* __restrict__ bias_hv,
    const unsigned short* __restrict__ Tke, const unsigned short* __restrict__ Tk2,
    const unsigned short* __restrict__ Tve, const unsigned short* __restrict__ Tv2,
    const float* __restrict__ k2b, const float* __restrict__ v2b,
    float* __restrict__ a_buf, float* __restrict__ v_buf) {

    __shared__ unsigned short sE[64][LDW];
    __shared__ unsigned short sH[64][LDW];
    __shared__ int sIdx[128];                 // [0..63]=src, [64..127]=dst

    const int tid   = threadIdx.x;
    const int lane  = tid & 31;
    const int wave  = tid >> 5;
    const int rowg  = (wave & 3) << 4;
    const int tbase = (wave >> 2) * 4;
    const int colg  = tbase * 16;
    const long long e0 = (long long)blockIdx.x * 64;
    const float INVS = 0.08838834764831845f;  // 1/sqrt(128)

    if (tid < 64) {
        long long e = e0 + tid; if (e >= E) e = E - 1;
        sIdx[tid]      = eidx[e];
        sIdx[64 + tid] = eidx[(long long)E + e];
    }
    #pragma unroll
    for (int i = tid; i < 64 * 32; i += 256) {
        int r = i >> 5, c = (i & 31) << 2;
        long long row = e0 + r; if (row >= E) row = E - 1;
        float4 f = *(const float4*)(edgef + row * D + c);
        uint2 p;
        p.x = ((unsigned int)f2bf(f.y) << 16) | f2bf(f.x);
        p.y = ((unsigned int)f2bf(f.w) << 16) | f2bf(f.z);
        *(uint2*)&sE[r][c] = p;
    }
    __syncthreads();

    const int mb = rowg + ((lane >> 4) << 3);
    f32x8 acc[4];

    // ---- K branch: GEMM1 (E @ Wke) + gathered init, silu -> sH ----
    wmma_quad(&sE[0][0], Tke, rowg, tbase, lane, acc);
    #pragma unroll
    for (int j = 0; j < 4; ++j) {
        int n = colg + j * 16 + (lane & 15);
        const float bh = bias_hk[n];
        #pragma unroll
        for (int r = 0; r < 8; ++r) {
            int m = mb + r;
            long long s = sIdx[m], dd = sIdx[64 + m];
            float h = acc[j][r] + Ka[s * D + n] + Kb[dd * D + n] + bh;
            sH[m][n] = f2bf(silu_f(h));
        }
    }
    __syncthreads();

    // ---- K branch: GEMM2 (silu @ k2), a = q*k/scale ----
    wmma_quad(&sH[0][0], Tk2, rowg, tbase, lane, acc);
    #pragma unroll
    for (int j = 0; j < 4; ++j) {
        int n = colg + j * 16 + (lane & 15);
        const float kb = k2b[n];
        #pragma unroll
        for (int r = 0; r < 8; ++r) {
            int m = mb + r;
            long long row = e0 + m;
            long long dd = sIdx[64 + m];
            float q = Qa[dd * D + n];
            if (row < E) a_buf[row * D + n] = q * (acc[j][r] + kb) * INVS;
        }
    }
    __syncthreads();   // all sH readers done before V branch overwrites it

    // ---- V branch: GEMM1 (E @ Wve) + gathered init, silu -> sH ----
    wmma_quad(&sE[0][0], Tve, rowg, tbase, lane, acc);
    #pragma unroll
    for (int j = 0; j < 4; ++j) {
        int n = colg + j * 16 + (lane & 15);
        const float bh = bias_hv[n];
        #pragma unroll
        for (int r = 0; r < 8; ++r) {
            int m = mb + r;
            long long s = sIdx[m], dd = sIdx[64 + m];
            float h = acc[j][r] + Va[s * D + n] + Vb[dd * D + n] + bh;
            sH[m][n] = f2bf(silu_f(h));
        }
    }
    __syncthreads();

    // ---- V branch: GEMM2 (silu @ v2) -> v ----
    wmma_quad(&sH[0][0], Tv2, rowg, tbase, lane, acc);
    #pragma unroll
    for (int j = 0; j < 4; ++j) {
        int n = colg + j * 16 + (lane & 15);
        const float vb = v2b[n];
        #pragma unroll
        for (int r = 0; r < 8; ++r) {
            long long row = e0 + mb + r;
            if (row < E) v_buf[row * D + n] = acc[j][r] + vb;
        }
    }
}

// ================= BN / scatter chain =================

__global__ void zero_kernel(float* __restrict__ p, long long n) {
    long long i  = (long long)blockIdx.x * blockDim.x + threadIdx.x;
    long long st = (long long)gridDim.x * blockDim.x;
    for (; i < n; i += st) p[i] = 0.0f;
}

// per-feature sum/sumsq; grid*block is a multiple of 128 so each thread owns one column
__global__ void colsum_kernel(const float* __restrict__ x, long long n,
                              float* __restrict__ sum, float* __restrict__ sq) {
    long long i0 = (long long)blockIdx.x * blockDim.x + threadIdx.x;
    long long st = (long long)gridDim.x * blockDim.x;
    int c = (int)(i0 & 127);
    float s = 0.0f, s2 = 0.0f;
    for (long long i = i0; i < n; i += st) { float v = x[i]; s += v; s2 += v * v; }
    atomicAdd(sum + c, s);
    atomicAdd(sq + c, s2);
}

__global__ void finalize_kernel(const float* __restrict__ sum, const float* __restrict__ sq,
                                float invE, float* __restrict__ mean, float* __restrict__ istd) {
    int c = threadIdx.x;
    float mu  = sum[c] * invE;
    float var = sq[c] * invE - mu * mu;
    mean[c] = mu;
    istd[c] = rsqrtf(var + 1e-5f);
}

// m = sigmoid(BN1(a)) * v, in place over a_buf; accumulates BN2 stats
__global__ void bn_sig_kernel(float* __restrict__ a, const float* __restrict__ v,
                              const float* __restrict__ mean, const float* __restrict__ istd,
                              const float* __restrict__ gamma, const float* __restrict__ beta,
                              float* __restrict__ sum2, float* __restrict__ sq2, long long n) {
    long long i0 = (long long)blockIdx.x * blockDim.x + threadIdx.x;
    long long st = (long long)gridDim.x * blockDim.x;
    int c = (int)(i0 & 127);
    const float g = gamma[c], b = beta[c], mu = mean[c], is = istd[c];
    float s = 0.0f, s2 = 0.0f;
    for (long long i = i0; i < n; i += st) {
        float al = g * (a[i] - mu) * is + b;
        float sg = 1.0f / (1.0f + __expf(-al));
        float m  = sg * v[i];
        a[i] = m;
        s += m; s2 += m * m;
    }
    atomicAdd(sum2 + c, s);
    atomicAdd(sq2 + c, s2);
}

// message = BN2(m); edge_out = softplus(edge + message); agg[dst] += message
__global__ void msg_scatter_kernel(const float* __restrict__ m_, const float* __restrict__ edgef,
                                   const int* __restrict__ eidx, int E,
                                   const float* __restrict__ mean, const float* __restrict__ istd,
                                   const float* __restrict__ gamma, const float* __restrict__ beta,
                                   float* __restrict__ agg, float* __restrict__ edge_out, long long n) {
    long long i0 = (long long)blockIdx.x * blockDim.x + threadIdx.x;
    long long st = (long long)gridDim.x * blockDim.x;
    for (long long i = i0; i < n; i += st) {
        int c = (int)(i & 127);
        long long e = i >> 7;
        float msg = gamma[c] * (m_[i] - mean[c]) * istd[c] + beta[c];
        atomicAdd(&agg[(long long)eidx[(long long)E + e] * D + c], msg);
        float x = edgef[i] + msg;
        edge_out[i] = (x > 20.0f) ? x : log1pf(__expf(x));
    }
}

__global__ void atom_out_kernel(const float* __restrict__ atomf, const float* __restrict__ agg,
                                float* __restrict__ out, long long n) {
    long long i = (long long)blockIdx.x * blockDim.x + threadIdx.x;
    if (i < n) out[i] = atomf[i] + agg[i];
}

// ================= host =================

extern "C" void kernel_launch(void* const* d_in, const int* in_sizes, int n_in,
                              void* d_out, int out_size, void* d_ws, size_t ws_size,
                              hipStream_t stream) {
    const float* atomf = (const float*)d_in[0];
    const float* edgef = (const float*)d_in[1];
    const int*   eidx  = (const int*)d_in[2];
    const float* Wq  = (const float*)d_in[3];  const float* bq  = (const float*)d_in[4];
    const float* Wk  = (const float*)d_in[5];  const float* bk  = (const float*)d_in[6];
    const float* Wv  = (const float*)d_in[7];  const float* bv  = (const float*)d_in[8];
    const float* We  = (const float*)d_in[9];  const float* be  = (const float*)d_in[10];
    const float* k1w = (const float*)d_in[11]; const float* k1b = (const float*)d_in[12];
    const float* k2w = (const float*)d_in[13]; const float* k2b = (const float*)d_in[14];
    const float* v1w = (const float*)d_in[15]; const float* v1b = (const float*)d_in[16];
    const float* v2w = (const float*)d_in[17]; const float* v2b = (const float*)d_in[18];
    const float* g1  = (const float*)d_in[19]; const float* b1  = (const float*)d_in[20];
    const float* g2  = (const float*)d_in[21]; const float* b2  = (const float*)d_in[22];

    const int N = in_sizes[0] / D;
    const int E = in_sizes[1] / D;
    const long long nN = (long long)N * D;
    const long long nE = (long long)E * D;

    // workspace carve-out
    char* ws = (char*)d_ws;
    size_t off = 0;
    auto carve = [&](size_t bytes) { void* p = ws + off; off = (off + bytes + 255) & ~(size_t)255; return p; };
    unsigned short* Tcomb = (unsigned short*)carve((size_t)6 * D * D * 2);
    unsigned short* Ttr   = (unsigned short*)carve((size_t)3 * D * D * 2);
    float* bias_hk = (float*)carve(D * 4);
    float* bias_hv = (float*)carve(D * 4);
    float* stats   = (float*)carve(1024 * 4);
    float* Qa  = (float*)carve(nN * 4);
    float* Ka  = (float*)carve(nN * 4);
    float* Kb  = (float*)carve(nN * 4);
    float* Va  = (float*)carve(nN * 4);
    float* Vb  = (float*)carve(nN * 4);
    float* agg = (float*)carve(nN * 4);
    float* a_buf = (float*)carve(nE * 4);
    float* v_buf = (float*)carve(nE * 4);

    float* sum1  = stats;        float* sq1   = stats + 128;
    float* sum2  = stats + 256;  float* sq2   = stats + 384;
    float* mean1 = stats + 512;  float* istd1 = stats + 640;
    float* mean2 = stats + 768;  float* istd2 = stats + 896;

    float* out_atom = (float*)d_out;
    float* out_edge = out_atom + nN;

    // fresh accumulators each call (deterministic)
    zero_kernel<<<4, 256, 0, stream>>>(stats, 512);
    zero_kernel<<<1024, 256, 0, stream>>>(agg, nN);

    // weight folding / fragment packing / bias folding (tiny)
    prep_combined_kernel<<<dim3(6, 128), 128, 0, stream>>>(Wk, Wv, We, k1w, v1w, Tcomb);
    prep_transpose_kernel<<<dim3(3, 128), 128, 0, stream>>>(Wq, k2w, v2w, Ttr);
    prep_bias_kernel<<<1, 128, 0, stream>>>(bk, be, k1w, k1b, bv, v1w, v1b, bias_hk, bias_hv);

    // atom-stage projections (L2-resident results)
    const int gN = (N + 63) / 64;
    gemm_proj_kernel<<<gN, 256, 0, stream>>>(atomf, N, Ttr,                bq,      Qa);
    gemm_proj_kernel<<<gN, 256, 0, stream>>>(atomf, N, Tcomb + 0 * 16384, nullptr, Ka);
    gemm_proj_kernel<<<gN, 256, 0, stream>>>(atomf, N, Tcomb + 1 * 16384, nullptr, Kb);
    gemm_proj_kernel<<<gN, 256, 0, stream>>>(atomf, N, Tcomb + 3 * 16384, nullptr, Va);
    gemm_proj_kernel<<<gN, 256, 0, stream>>>(atomf, N, Tcomb + 4 * 16384, nullptr, Vb);

    // fused edge stage: 4 back-to-back WMMA GEMMs per 64-edge tile, B operands from L2
    const int gE = (E + 63) / 64;
    edge_fused_kernel<<<gE, 256, 0, stream>>>(edgef, eidx, E, Qa, Ka, Kb, Va, Vb,
                                              bias_hk, bias_hv,
                                              Tcomb + 2 * 16384, Ttr + 1 * 16384,
                                              Tcomb + 5 * 16384, Ttr + 2 * 16384,
                                              k2b, v2b, a_buf, v_buf);

    const float invE = 1.0f / (float)E;
    // BN1 stats -> alpha -> sigmoid*v (+BN2 stats) -> BN2 -> scatter/softplus
    colsum_kernel<<<1024, 256, 0, stream>>>(a_buf, nE, sum1, sq1);
    finalize_kernel<<<1, 128, 0, stream>>>(sum1, sq1, invE, mean1, istd1);
    bn_sig_kernel<<<2048, 256, 0, stream>>>(a_buf, v_buf, mean1, istd1, g1, b1, sum2, sq2, nE);
    finalize_kernel<<<1, 128, 0, stream>>>(sum2, sq2, invE, mean2, istd2);
    msg_scatter_kernel<<<2048, 256, 0, stream>>>(a_buf, edgef, eidx, E, mean2, istd2, g2, b2,
                                                 agg, out_edge, nE);
    atom_out_kernel<<<(unsigned)((nN + 255) / 256), 256, 0, stream>>>(atomf, agg, out_atom, nN);
}
